// LegacyCompatiblePointerAttention_41162966565292
// MI455X (gfx1250) — compile-verified, matrix operand order checked by reading, not run on hardware
//
#include <hip/hip_runtime.h>
#include <hip/hip_bf16.h>
#include <stdint.h>

// Problem constants (match reference)
#define BB   512
#define NN   1000
#define DD   128
#define HH   8
#define HIDN 128
#define NPAD 1024
#define CTX_STRIDE_B 272   // bytes per padded bf16 ctx row (136 bf16 = 128 + 8 pad)

// ---- LDS layout (bytes) -----------------------------------------------------
#define OFF_CTX    0
#define OFF_COMP   (NPAD * CTX_STRIDE_B)      // 278528: comp f32[8][1024] / attn bf16[16][1024] / u f32[1024]
#define OFF_GT     (OFF_COMP + 32768)         // 311296: gT  bf16 [16][128]
#define OFF_W2T    (OFF_GT  + 4096)           // 315392: w2T bf16 [16][128]
#define OFF_STATE  (OFF_W2T + 4096)           // 319488: 384 f32
#define OFF_Q      (OFF_STATE + 1536)         // 321024: 128 f32
#define OFF_CBAR   (OFF_Q + 512)              // 321536: 8x128 f32
#define OFF_X      (OFF_CBAR + 4096)          // 325632
#define OFF_Y      (OFF_X + 512)              // 326144
#define OFF_RED    (OFF_Y + 512)              // 326656
#define LDS_BYTES  (OFF_RED + 128)            // 326784 <= 327680 (320 KB / WGP)

typedef __bf16 v16bf __attribute__((ext_vector_type(16)));
typedef float  v8f   __attribute__((ext_vector_type(8)));

union FragBF { uint4 q[2]; uint32_t u[8]; v16bf v; };

__device__ __forceinline__ unsigned short f2bf(float f) {   // RNE f32 -> bf16
  unsigned u = __float_as_uint(f);
  u += 0x7FFFu + ((u >> 16) & 1u);
  return (unsigned short)(u >> 16);
}

// 16-bit A/B fragment loader for 16x16x32 WMMA from a row-major bf16 LDS tile.
// A: lane = M row, regs hold K pairs. B: lane = N col, regs hold K pairs.
// v0..3 : K = kb+{0..7} (lanes 0-15) / kb+{8..15} (lanes 16-31)
// v4..7 : K = kb+{16..23} / kb+{24..31}
__device__ __forceinline__ v16bf load_frag(const char* base, int rowStrideB,
                                           int row0, int kb, int lane) {
  int r  = row0 + (lane & 15);
  int hi = (lane >> 4) & 1;
  const char* p = base + r * rowStrideB + (kb + hi * 8) * 2;
  FragBF f;
  f.q[0] = *(const uint4*)(p);
  f.q[1] = *(const uint4*)(p + 32);
  return f.v;
}

__device__ __forceinline__ float wave_max(float v) {
  #pragma unroll
  for (int d = 16; d; d >>= 1) v = fmaxf(v, __shfl_xor(v, d, 32));
  return v;
}
__device__ __forceinline__ float wave_sum(float v) {
  #pragma unroll
  for (int d = 16; d; d >>= 1) v += __shfl_xor(v, d, 32);
  return v;
}

__global__ __launch_bounds__(256)
void pointer_attn_kernel(const float* __restrict__ state_t,
                         const float* __restrict__ context,
                         const int*   __restrict__ mask,
                         const float* __restrict__ Wq,
                         const float* __restrict__ Wk_mha,
                         const float* __restrict__ Wv,
                         const float* __restrict__ Wfc,
                         const float* __restrict__ Wk,
                         const int*   __restrict__ Tp,
                         float* __restrict__ out) {
  extern __shared__ char smem[];
  const int b    = blockIdx.x;
  const int tid  = threadIdx.x;
  const int wave = tid >> 5;
  const int lane = tid & 31;

  float* stateL = (float*)(smem + OFF_STATE);
  float* qL     = (float*)(smem + OFF_Q);
  float* cbarL  = (float*)(smem + OFF_CBAR);
  float* xL     = (float*)(smem + OFF_X);
  float* yL     = (float*)(smem + OFF_Y);
  float* redL   = (float*)(smem + OFF_RED);
  const int* maskrow = mask + b * NN;

  // ---- Phase A: state -> LDS; zero gT / w2T pad regions ---------------------
  for (int i = tid; i < 3 * DD; i += 256) stateL[i] = state_t[b * 3 * DD + i];
  *(uint4*)(smem + OFF_GT  + tid * 16) = make_uint4(0, 0, 0, 0);
  *(uint4*)(smem + OFF_W2T + tid * 16) = make_uint4(0, 0, 0, 0);
  __syncthreads();

  // q[j] = state · Wq[:,j]   (384 MACs x 128 threads; Wq coalesced over j)
  if (tid < HIDN) {
    float acc = 0.f;
    for (int i = 0; i < 3 * DD; ++i) acc += stateL[i] * Wq[i * HIDN + tid];
    qL[tid] = acc;
  }
  __syncthreads();

  // ---- Phase B: gT[h][c] = (1/sqrt(16)) * sum_d Wk_mha[c][16h+d]*q[16h+d] ---
  for (int idx = tid; idx < DD * HH; idx += 256) {
    int c = idx >> 3, h = idx & 7;
    const float* wrow = Wk_mha + c * HIDN + h * 16;
    const float* qh   = qL + h * 16;
    float acc = 0.f;
    #pragma unroll
    for (int d = 0; d < 16; ++d) acc += wrow[d] * qh[d];
    *(unsigned short*)(smem + OFF_GT + h * 256 + c * 2) = f2bf(acc * 0.25f);
  }

  // ---- Phase C-load: the ONLY HBM pass over context (f32 -> bf16 LDS) -------
  const float4* ctx4 = (const float4*)context;
  for (int i = tid; i < NPAD * 32; i += 256) {
    int n = i >> 5, c4 = i & 31;
    float4 f = make_float4(0.f, 0.f, 0.f, 0.f);
    if (n < NN) f = ctx4[(size_t)(b * NN + n) * 32 + c4];
    uint2 p;
    p.x = (unsigned)f2bf(f.x) | ((unsigned)f2bf(f.y) << 16);
    p.y = (unsigned)f2bf(f.z) | ((unsigned)f2bf(f.w) << 16);
    *(uint2*)(smem + OFF_CTX + n * CTX_STRIDE_B + c4 * 8) = p;
  }
  __syncthreads();

  // ---- Phase C: comp[n,h] = ctx(N x 128) @ g(128 x 8pad16)   [WMMA] ---------
  {
    v16bf bg[4];
    #pragma unroll
    for (int t = 0; t < 4; ++t) bg[t] = load_frag(smem + OFF_GT, 256, 0, 32 * t, lane);
    for (int r = wave; r < NPAD / 16; r += 8) {
      v8f acc = {0.f, 0.f, 0.f, 0.f, 0.f, 0.f, 0.f, 0.f};
      #pragma unroll
      for (int t = 0; t < 4; ++t) {
        v16bf a = load_frag(smem + OFF_CTX, CTX_STRIDE_B, 16 * r, 32 * t, lane);
        acc = __builtin_amdgcn_wmma_f32_16x16x32_bf16(false, a, false, bg[t],
                                                      (short)0, acc, false, false);
      }
      if ((lane & 15) < 8) {                 // valid head columns only
        int h  = lane & 15;
        int n0 = 16 * r + ((lane >> 4) ? 8 : 0);
        #pragma unroll
        for (int g = 0; g < 8; ++g)
          *(float*)(smem + OFF_COMP + h * 4096 + (n0 + g) * 4) = acc[g];
      }
    }
  }
  __syncthreads();

  // ---- Phase D: per-head masked softmax (wave h owns head h) ----------------
  {
    float vals[32];
    float m = -1e30f;
    for (int i = 0; i < 32; ++i) {
      int n = lane + 32 * i;
      float cv = *(const float*)(smem + OFF_COMP + wave * 4096 + n * 4);
      bool valid = (n < NN) && (maskrow[n] == 0);
      float v = valid ? cv : -1e30f;
      vals[i] = v;
      m = fmaxf(m, v);
    }
    m = wave_max(m);
    float s = 0.f;
    for (int i = 0; i < 32; ++i) {
      float e = (vals[i] <= -1e30f) ? 0.f : __expf(vals[i] - m);
      vals[i] = e; s += e;
    }
    s = wave_sum(s);
    float inv = 1.f / s;
    __syncthreads();                        // all waves done reading comp
    for (int i = 0; i < 32; ++i) {          // attn bf16 overwrites comp region
      int n = lane + 32 * i;
      *(unsigned short*)(smem + OFF_COMP + wave * 2048 + n * 2) = f2bf(vals[i] * inv);
    }
    #pragma unroll
    for (int j = 0; j < 8; ++j)             // zero pad heads 8..15
      *(unsigned long long*)(smem + OFF_COMP + (8 + wave) * 2048 + lane * 8 + j * 256) = 0ull;
  }
  __syncthreads();

  // ---- Phase E: cbar(8pad16 x 128) = attn(16 x N) @ ctx(N x 128)  [WMMA] ----
  {
    v8f acc = {0.f, 0.f, 0.f, 0.f, 0.f, 0.f, 0.f, 0.f};
    int ccol = 16 * wave + (lane & 15);
    int hi   = (lane >> 4) & 1;
    for (int kb = 0; kb < NPAD; kb += 32) {
      v16bf a = load_frag(smem + OFF_COMP, 2048, 0, kb, lane);  // attn rows
      FragBF bf;                                                // gather ctx cols
      #pragma unroll
      for (int v = 0; v < 4; ++v) {
        int k0 = kb + hi * 8 + 2 * v;
        unsigned lo = *(const unsigned short*)(smem + OFF_CTX + k0 * CTX_STRIDE_B + ccol * 2);
        unsigned h2 = *(const unsigned short*)(smem + OFF_CTX + (k0 + 1) * CTX_STRIDE_B + ccol * 2);
        bf.u[v] = lo | (h2 << 16);
      }
      #pragma unroll
      for (int v = 0; v < 4; ++v) {
        int k0 = kb + 16 + hi * 8 + 2 * v;
        unsigned lo = *(const unsigned short*)(smem + OFF_CTX + k0 * CTX_STRIDE_B + ccol * 2);
        unsigned h2 = *(const unsigned short*)(smem + OFF_CTX + (k0 + 1) * CTX_STRIDE_B + ccol * 2);
        bf.u[4 + v] = lo | (h2 << 16);
      }
      acc = __builtin_amdgcn_wmma_f32_16x16x32_bf16(false, a, false, bf.v,
                                                    (short)0, acc, false, false);
    }
    if (lane < 16) {                         // rows 0..7 = heads
      #pragma unroll
      for (int g = 0; g < 8; ++g)
        *(float*)(smem + OFF_CBAR + g * 512 + (16 * wave + lane) * 4) = acc[g];
    }
  }
  __syncthreads();

  // ---- Phase F: x = cbar·Wv ; y = x@Wfc ; w2 = (1/sqrt(128))·Wk@y -----------
  if (tid < HIDN) {
    int h = tid >> 4;
    float acc = 0.f;
    for (int c = 0; c < DD; ++c) acc += cbarL[h * 128 + c] * Wv[c * HIDN + tid];
    xL[tid] = acc;
  }
  __syncthreads();
  if (tid < HIDN) {
    float acc = 0.f;
    for (int c = 0; c < HIDN; ++c) acc += xL[c] * Wfc[c * HIDN + tid];
    yL[tid] = acc;
  }
  __syncthreads();
  if (tid < DD) {
    const float* wr = Wk + tid * HIDN;
    float acc = 0.f;
    for (int j = 0; j < HIDN; ++j) acc += wr[j] * yL[j];
    *(unsigned short*)(smem + OFF_W2T + tid * 2) = f2bf(acc * 0.08838834764831845f);
  }
  __syncthreads();

  // ---- Phase G: compat[n] = ctx(N x 128) @ w2(128 x 1pad16)   [WMMA] --------
  {
    v16bf bw[4];
    #pragma unroll
    for (int t = 0; t < 4; ++t) bw[t] = load_frag(smem + OFF_W2T, 256, 0, 32 * t, lane);
    for (int r = wave; r < NPAD / 16; r += 8) {
      v8f acc = {0.f, 0.f, 0.f, 0.f, 0.f, 0.f, 0.f, 0.f};
      #pragma unroll
      for (int t = 0; t < 4; ++t) {
        v16bf a = load_frag(smem + OFF_CTX, CTX_STRIDE_B, 16 * r, 32 * t, lane);
        acc = __builtin_amdgcn_wmma_f32_16x16x32_bf16(false, a, false, bw[t],
                                                      (short)0, acc, false, false);
      }
      if ((lane & 15) == 0) {                // column 0 holds compat
        int n0 = 16 * r + ((lane >> 4) ? 8 : 0);
        #pragma unroll
        for (int g = 0; g < 8; ++g)
          *(float*)(smem + OFF_COMP + (n0 + g) * 4) = acc[g];   // u[] reuse
      }
    }
  }
  __syncthreads();

  // ---- Phase H: u = 5*tanh(compat); mask; max-sub; temperature softmax ------
  float invT = 1.0f / (float)(*Tp);
  float uv[4];
  float m = -1e30f;
  #pragma unroll
  for (int i = 0; i < 4; ++i) {
    int n = tid + 256 * i;
    float v = -1e30f;
    if (n < NN && maskrow[n] == 0) {
      float raw = *(const float*)(smem + OFF_COMP + n * 4);
      v = 5.0f * tanhf(raw);
    }
    uv[i] = v;
    m = fmaxf(m, v);
  }
  m = wave_max(m);
  if (lane == 0) redL[wave] = m;
  __syncthreads();
  float mm = redL[0];
  #pragma unroll
  for (int j = 1; j < 8; ++j) mm = fmaxf(mm, redL[j]);
  float s = 0.f;
  #pragma unroll
  for (int i = 0; i < 4; ++i) {
    float e = (uv[i] <= -1e30f) ? 0.f : __expf((uv[i] - mm) * invT);
    uv[i] = e; s += e;
  }
  s = wave_sum(s);
  __syncthreads();
  if (lane == 0) redL[wave] = s;
  __syncthreads();
  float ss = 0.f;
  #pragma unroll
  for (int j = 0; j < 8; ++j) ss += redL[j];
  float invS = 1.f / ss;
  #pragma unroll
  for (int i = 0; i < 4; ++i) {
    int n = tid + 256 * i;
    if (n < NN) out[(size_t)b * NN + n] = uv[i] * invS;
  }
}

extern "C" void kernel_launch(void* const* d_in, const int* in_sizes, int n_in,
                              void* d_out, int out_size, void* d_ws, size_t ws_size,
                              hipStream_t stream) {
  (void)in_sizes; (void)n_in; (void)out_size; (void)d_ws; (void)ws_size;
  const float* state_t = (const float*)d_in[0];
  const float* context = (const float*)d_in[1];
  const int*   mask    = (const int*)d_in[2];
  const float* Wq      = (const float*)d_in[3];
  const float* Wk_mha  = (const float*)d_in[4];
  const float* Wv      = (const float*)d_in[5];
  const float* Wfc     = (const float*)d_in[6];
  const float* Wk      = (const float*)d_in[7];
  const int*   Tp      = (const int*)d_in[8];
  float* out = (float*)d_out;

  // Allow >64KB dynamic LDS (host-side attribute; not a stream op, capture-safe)
  hipFuncSetAttribute(reinterpret_cast<const void*>(pointer_attn_kernel),
                      hipFuncAttributeMaxDynamicSharedMemorySize, LDS_BYTES);

  pointer_attn_kernel<<<BB, 256, LDS_BYTES, stream>>>(
      state_t, context, mask, Wq, Wk_mha, Wv, Wfc, Wk, Tp, out);
}
